// MultiHeadAttn_50525995270430
// MI455X (gfx1250) — matrix operand order repaired and checked by reading, not compiled, not run on hardware
//
#include <hip/hip_runtime.h>
#include <hip/hip_bf16.h>

// ---------------------------------------------------------------------------
// MHA forward for MI455X (gfx1250, wave32, WMMA).
// b=4, t=4096, INPUT_DIM=768, HE=768, EMBED_DIM=64. ~266 GFLOP -> compute
// bound: bf16 v_wmma_f32_16x16x32_bf16 with fp32 accum, fp32 softmax.
// Attention (268 MB) is a required output -> materialize scores in d_out,
// softmax in place; 192 MB L2 absorbs Q/K/V and score-tile re-reads.
// ---------------------------------------------------------------------------

typedef __attribute__((ext_vector_type(16))) __bf16 v16bf;
typedef __attribute__((ext_vector_type(8)))  __bf16 v8bf;
typedef __attribute__((ext_vector_type(2)))  __bf16 v2bf;
typedef __attribute__((ext_vector_type(8)))  float  v8f;

// ---- fp32 -> bf16 (pair-packed, RNE) --------------------------------------
static __device__ inline unsigned bf1(float f) {   // 16-bit result in low bits
  unsigned u = __builtin_bit_cast(unsigned, f);
  u += 0x7FFFu + ((u >> 16) & 1u);
  return u >> 16;
}
static __device__ inline unsigned pack2(float a, float b) {
#if __has_builtin(__builtin_amdgcn_cvt_pk_bf16_f32)
  v2bf r = __builtin_amdgcn_cvt_pk_bf16_f32(a, b);   // 1 VALU op
  return __builtin_bit_cast(unsigned, r);
#else
  return bf1(a) | (bf1(b) << 16);
#endif
}
static __device__ inline unsigned short f2bfu(float f) {
  return (unsigned short)bf1(f);
}

// ---- WMMA fragment loaders (layouts per ISA 7.12.2) -----------------------
// A 16x32 bf16: lane L -> row L%16; elems 0..7 = K kb..kb+7,
// elems 8..15 = K kb+16..kb+23, kb = kk + (L/16)*8.
static __device__ inline v16bf load_a_bf16(const __bf16* A, int lda, int row,
                                           int kk, int lane) {
  const __bf16* p = A + (long)row * lda + kk + ((lane >> 4) << 3);
  v8bf lo = *(const v8bf*)p;
  v8bf hi = *(const v8bf*)(p + 16);
  return __builtin_shufflevector(lo, hi, 0, 1, 2, 3, 4, 5, 6, 7,
                                 8, 9, 10, 11, 12, 13, 14, 15);
}
static __device__ inline v16bf load_a_f32(const float* A, int lda, int row,
                                          int kk, int lane) {
  const float* p = A + (long)row * lda + kk + ((lane >> 4) << 3);
  union { v16bf v; unsigned u[8]; } r;
#pragma unroll
  for (int i = 0; i < 4; ++i) {
    r.u[i]     = pack2(p[2 * i],      p[2 * i + 1]);
    r.u[4 + i] = pack2(p[16 + 2 * i], p[16 + 2 * i + 1]);
  }
  return r.v;
}
// B 32x16 bf16: lane L -> col L%16; elems 0..7 = K kb..kb+7,
// elems 8..15 = K kb+16..kb+23, kb = kk + (L/16)*8 (row n of Bt[N,K]).
static __device__ inline v16bf load_b_bf16(const __bf16* Bt, int ldb, int n,
                                           int kk, int lane) {
  const __bf16* p = Bt + (long)n * ldb + kk + ((lane >> 4) << 3);
  v8bf lo = *(const v8bf*)p;
  v8bf hi = *(const v8bf*)(p + 16);
  return __builtin_shufflevector(lo, hi, 0, 1, 2, 3, 4, 5, 6, 7,
                                 8, 9, 10, 11, 12, 13, 14, 15);
}

// ---- Wave-level GEMM core: 32(M) x 64(N) tile, K-step 32, 8 WMMA/step -----
template <bool AF32>
static __device__ inline void gemm_core(const void* Ap, int lda,
                                        const __bf16* Bt, int ldb,
                                        int m0, int n0, int Nmax, int K,
                                        int lane, v8f (&acc)[2][4]) {
  for (int kk = 0; kk < K; kk += 32) {
    v16bf a[2], b[4];
#pragma unroll
    for (int i = 0; i < 2; ++i) {
      int row = m0 + 16 * i + (lane & 15);
      if constexpr (AF32) a[i] = load_a_f32((const float*)Ap, lda, row, kk, lane);
      else                a[i] = load_a_bf16((const __bf16*)Ap, lda, row, kk, lane);
    }
#pragma unroll
    for (int j = 0; j < 4; ++j) {
      int n = n0 + 16 * j + (lane & 15);
      n = n < Nmax ? n : Nmax - 1;  // clamp (reads stay in-bounds)
      b[j] = load_b_bf16(Bt, ldb, n, kk, lane);
    }
#pragma unroll
    for (int i = 0; i < 2; ++i)
#pragma unroll
      for (int j = 0; j < 4; ++j)
        acc[i][j] = __builtin_amdgcn_wmma_f32_16x16x32_bf16(
            false, a[i], false, b[j], (short)0, acc[i][j], false, false);
  }
}
// C coords (ISA 7.12.2): VGPR v -> M = m0+16*i + 8*(L/16) + v, N = n0+16*j + L%16.

// ---- Kernel 0: elementwise fp32 -> bf16 (x, Wqkv, Wo pre-conversion) ------
__global__ __launch_bounds__(256) void cvt4_kernel(
    const float* __restrict__ s, unsigned short* __restrict__ d, long n4) {
  long i = (long)blockIdx.x * 256 + threadIdx.x;
  if (i >= n4) return;
  float4 f = ((const float4*)s)[i];
  uint2 o;
  o.x = pack2(f.x, f.y);
  o.y = pack2(f.z, f.w);
  ((uint2*)d)[i] = o;
}

// ---- Kernel 1: QKV projection + de-interleave (all-bf16 GEMM) -------------
// qkv[t][3d+w] = sum_i x[t][i]*Wqkv[3d+w][i] + bqkv[3d+w]
// Bt view: base = Wqkvb + w*768, ldb = 2304 -> row d hits Wqkv row 3d+w.
__global__ __launch_bounds__(256) void qkv_kernel(
    const unsigned short* __restrict__ xb, const unsigned short* __restrict__ Wqkvb,
    const float* __restrict__ bqkv, unsigned short* __restrict__ Q,
    unsigned short* __restrict__ Kb, unsigned short* __restrict__ Vt) {
  const int lane = threadIdx.x & 31, wave = threadIdx.x >> 5;
  const int m0 = blockIdx.y * 128 + (wave & 3) * 32;
  const int n0 = blockIdx.x * 128 + (wave >> 2) * 64;
  const int w  = blockIdx.z;
  v8f acc[2][4] = {};
  gemm_core<false>((const __bf16*)xb, 768, (const __bf16*)Wqkvb + w * 768, 2304,
                   m0, n0, 768, 768, lane, acc);
#pragma unroll
  for (int i = 0; i < 2; ++i)
#pragma unroll
    for (int j = 0; j < 4; ++j)
#pragma unroll
      for (int v = 0; v < 8; ++v) {
        int gm = m0 + 16 * i + ((lane >> 4) << 3) + v;
        int gn = n0 + 16 * j + (lane & 15);
        unsigned short bv = f2bfu(acc[i][j][v] + bqkv[3 * gn + w]);
        if (w == 0)      Q[(long)gm * 768 + gn] = bv;
        else if (w == 1) Kb[(long)gm * 768 + gn] = bv;
        else {           // store V transposed: Vt[b][d][t]
          int bb = gm >> 12, t = gm & 4095;
          Vt[(long)bb * 768 * 4096 + (long)gn * 4096 + t] = bv;
        }
      }
}

// ---- Kernel 2: scores = Q K^T * scale -> attention buffer (fp32) ----------
__global__ __launch_bounds__(256) void scores_kernel(
    const unsigned short* __restrict__ Q, const unsigned short* __restrict__ Kb,
    float* __restrict__ attn) {
  const int lane = threadIdx.x & 31, wave = threadIdx.x >> 5;
  const int m0 = blockIdx.y * 128 + (wave & 3) * 32;
  const int n0 = blockIdx.x * 128 + (wave >> 2) * 64;
  const int b  = blockIdx.z;
  const __bf16* A  = (const __bf16*)Q  + (long)b * 4096 * 768;
  const __bf16* Bt = (const __bf16*)Kb + (long)b * 4096 * 768;
  v8f acc[2][4] = {};
  gemm_core<false>(A, 768, Bt, 768, m0, n0, 4096, 768, lane, acc);
  const float scale = 0.036084391824351615f;  // 1/sqrt(768)
  float* C = attn + (long)b * 4096 * 4096;
#pragma unroll
  for (int i = 0; i < 2; ++i)
#pragma unroll
    for (int j = 0; j < 4; ++j)
#pragma unroll
      for (int v = 0; v < 8; ++v) {
        int gm = m0 + 16 * i + ((lane >> 4) << 3) + v;
        int gn = n0 + 16 * j + (lane & 15);
        C[(long)gm * 4096 + gn] = acc[i][j][v] * scale;
      }
}

// ---- Kernel 3: row softmax in place (4096 elems/row) ----------------------
__global__ __launch_bounds__(256) void softmax_kernel(float* __restrict__ attn) {
  __shared__ float red[256];
  float* p = attn + (long)blockIdx.x * 4096;
  const int tid = threadIdx.x;
  float v[16];
  float mx = -3.402823466e38f;
#pragma unroll
  for (int i = 0; i < 16; ++i) { v[i] = p[tid + 256 * i]; mx = fmaxf(mx, v[i]); }
  red[tid] = mx; __syncthreads();
  for (int s = 128; s > 0; s >>= 1) {
    if (tid < s) red[tid] = fmaxf(red[tid], red[tid + s]);
    __syncthreads();
  }
  mx = red[0]; __syncthreads();
  float sum = 0.f;
#pragma unroll
  for (int i = 0; i < 16; ++i) { v[i] = __expf(v[i] - mx); sum += v[i]; }
  red[tid] = sum; __syncthreads();
  for (int s = 128; s > 0; s >>= 1) {
    if (tid < s) red[tid] += red[tid + s];
    __syncthreads();
  }
  const float inv = 1.0f / red[0];
#pragma unroll
  for (int i = 0; i < 16; ++i) p[tid + 256 * i] = v[i] * inv;
}

// ---- Kernel 4: values = P V (P fp32 -> bf16 via cvt_pk, Bt = V^T) ---------
__global__ __launch_bounds__(256) void pv_kernel(
    const float* __restrict__ attn, const unsigned short* __restrict__ Vt,
    unsigned short* __restrict__ vals) {
  const int lane = threadIdx.x & 31, wave = threadIdx.x >> 5;
  const int m0 = blockIdx.y * 128 + (wave & 3) * 32;
  const int n0 = blockIdx.x * 128 + (wave >> 2) * 64;
  const int b  = blockIdx.z;
  const float*  A  = attn + (long)b * 4096 * 4096;
  const __bf16* Bt = (const __bf16*)Vt + (long)b * 768 * 4096;
  v8f acc[2][4] = {};
  gemm_core<true>(A, 4096, Bt, 4096, m0, n0, 768, 4096, lane, acc);
#pragma unroll
  for (int i = 0; i < 2; ++i)
#pragma unroll
    for (int j = 0; j < 4; ++j)
#pragma unroll
      for (int v = 0; v < 8; ++v) {
        int gm = m0 + 16 * i + ((lane >> 4) << 3) + v;
        int gn = n0 + 16 * j + (lane & 15);
        vals[((long)b * 4096 + gm) * 768 + gn] = f2bfu(acc[i][j][v]);
      }
}

// ---- Kernel 5: output = values Wo^T + bo (8 M-waves x 1 N-wave) -----------
__global__ __launch_bounds__(256) void out_kernel(
    const unsigned short* __restrict__ vals, const unsigned short* __restrict__ Wob,
    const float* __restrict__ bo, float* __restrict__ out) {
  const int lane = threadIdx.x & 31, wave = threadIdx.x >> 5;
  const int m0 = blockIdx.y * 256 + wave * 32;   // block tile 256(M) x 64(N)
  const int n0 = 0;
  v8f acc[2][4] = {};
  gemm_core<false>((const __bf16*)vals, 768, (const __bf16*)Wob, 768,
                   m0, n0, 64, 768, lane, acc);
#pragma unroll
  for (int i = 0; i < 2; ++i)
#pragma unroll
    for (int j = 0; j < 4; ++j)
#pragma unroll
      for (int v = 0; v < 8; ++v) {
        int gm = m0 + 16 * i + ((lane >> 4) << 3) + v;
        int gn = 16 * j + (lane & 15);
        out[(long)gm * 64 + gn] = acc[i][j][v] + bo[gn];
      }
}

// ---------------------------------------------------------------------------
extern "C" void kernel_launch(void* const* d_in, const int* in_sizes, int n_in,
                              void* d_out, int out_size, void* d_ws,
                              size_t ws_size, hipStream_t stream) {
  const float* x    = (const float*)d_in[0];
  const float* Wqkv = (const float*)d_in[1];
  const float* bqkv = (const float*)d_in[2];
  const float* Wo   = (const float*)d_in[3];
  const float* bo   = (const float*)d_in[4];

  float* out  = (float*)d_out;                 // (4,4096,64) fp32
  float* attn = out + (size_t)16384 * 64;      // (4,4096,4096) fp32

  // Workspace (bf16/ushort), ~130 MB total:
  const size_t NX = (size_t)16384 * 768;       // 12.58M elems
  const size_t NW = (size_t)2304 * 768;        // 1.77M
  const size_t NO = (size_t)64 * 768;          // 49K
  unsigned short* xb    = (unsigned short*)d_ws;
  unsigned short* Wqkvb = xb    + NX;
  unsigned short* Wob   = Wqkvb + NW;
  unsigned short* Q     = Wob   + NO;
  unsigned short* Kb    = Q     + NX;
  unsigned short* Vt    = Kb    + NX;
  unsigned short* Vals  = Vt    + NX;

  dim3 blk(256);
  // 0) one-time fp32 -> bf16 conversions (4 floats / thread)
  cvt4_kernel<<<dim3((unsigned)(NX / 4 / 256)), blk, 0, stream>>>(x,    xb,    NX / 4);
  cvt4_kernel<<<dim3((unsigned)(NW / 4 / 256)), blk, 0, stream>>>(Wqkv, Wqkvb, NW / 4);
  cvt4_kernel<<<dim3((unsigned)(NO / 4 / 256)), blk, 0, stream>>>(Wo,   Wob,   NO / 4);
  // 1..5) pipeline
  qkv_kernel    <<<dim3(6, 128, 3), blk, 0, stream>>>(xb, Wqkvb, bqkv, Q, Kb, Vt);
  scores_kernel <<<dim3(32, 32, 4), blk, 0, stream>>>(Q, Kb, attn);
  softmax_kernel<<<dim3(16384),     blk, 0, stream>>>(attn);
  pv_kernel     <<<dim3(6, 32, 4),  blk, 0, stream>>>(attn, Vt, Vals);
  out_kernel    <<<dim3(1, 64, 1),  blk, 0, stream>>>(Vals, Wob, bo, out);
}
